// SelfAttention_55533927137696
// MI455X (gfx1250) — compile-verified, matrix operand order checked by reading, not run on hardware
//
#include <hip/hip_runtime.h>

// Problem constants (match reference).
#define BB 4
#define CC 256
#define NN 4096
#define DD 32

typedef __attribute__((ext_vector_type(16))) __bf16 v16bf;
typedef __attribute__((ext_vector_type(8)))  __bf16 v8bf;
typedef __attribute__((ext_vector_type(8)))  float  v8f;
typedef __attribute__((ext_vector_type(4)))  float  v4f;

static __device__ __forceinline__ v16bf cat8(v8bf lo, v8bf hi) {
  v16bf r;
#pragma unroll
  for (int i = 0; i < 8; ++i) { r[i] = lo[i]; r[8 + i] = hi[i]; }
  return r;
}

// -------------------------------------------------------------------------
// Kernel 1: q/k/v 1x1-conv projections.  x:[B,C,N] fp32 -> q,k:[B,N,32] bf16,
// vt:[B,32,N] bf16 (V transposed so WMMA B-fragments are contiguous loads).
// -------------------------------------------------------------------------
__global__ __launch_bounds__(256) void proj_qkv(
    const float* __restrict__ x,
    const float* __restrict__ Wq, const float* __restrict__ bq,
    const float* __restrict__ Wk, const float* __restrict__ bk,
    const float* __restrict__ Wv, const float* __restrict__ bv,
    __bf16* __restrict__ qo, __bf16* __restrict__ ko, __bf16* __restrict__ vto)
{
  __shared__ float xt[CC][DD];          // 32 KB tile: all channels x 32 pixels
  const int b  = blockIdx.x / (NN / 32);
  const int n0 = (blockIdx.x % (NN / 32)) * 32;
  const int t  = threadIdx.x;

  {   // cooperative load: thread t loads channel row c==t, 32 pixels
    const float* src = x + ((size_t)(b * CC + t)) * NN + n0;
#pragma unroll
    for (int i = 0; i < 8; ++i)
      *(float4*)&xt[t][i * 4] = *(const float4*)&src[i * 4];
  }
  __syncthreads();

  const int nl = t & 31;                // pixel within tile
  const int dg = t >> 5;                // d group: handles d = 4*dg .. 4*dg+3
  float aq[4] = {0, 0, 0, 0}, ak[4] = {0, 0, 0, 0}, av[4] = {0, 0, 0, 0};
  for (int c = 0; c < CC; ++c) {
    const float xv = xt[c][nl];
#pragma unroll
    for (int i = 0; i < 4; ++i) {
      const int d = dg * 4 + i;
      aq[i] = fmaf(Wq[d * CC + c], xv, aq[i]);
      ak[i] = fmaf(Wk[d * CC + c], xv, ak[i]);
      av[i] = fmaf(Wv[d * CC + c], xv, av[i]);
    }
  }
  const int n = n0 + nl;
#pragma unroll
  for (int i = 0; i < 4; ++i) {
    const int d = dg * 4 + i;
    qo [((size_t)b * NN + n) * DD + d] = (__bf16)(aq[i] + bq[d]);
    ko [((size_t)b * NN + n) * DD + d] = (__bf16)(ak[i] + bk[d]);
    vto[((size_t)b * DD + d) * NN + n] = (__bf16)(av[i] + bv[d]);
  }
}

// -------------------------------------------------------------------------
// Kernel 2: attention.  One wave per 16-query block. Three streamed QK^T
// passes (row max -> row sum-exp -> normalize+write atten) + fused P@V^T.
// D=32 == bf16 WMMA K-dim, so each 16x16 S tile is exactly one
// v_wmma_f32_16x16x32_bf16.
// -------------------------------------------------------------------------
__global__ __launch_bounds__(128) void attn(
    const __bf16* __restrict__ qm, const __bf16* __restrict__ km,
    const __bf16* __restrict__ vtm,
    float* __restrict__ atten, float* __restrict__ obuf)
{
  __shared__ float ldsP[4][16][36];     // per-wave P staging, padded stride
  const int lane = threadIdx.x & 31;
  const int wave = threadIdx.x >> 5;
  const int half = lane >> 4;           // 16-lane half (ISA fragment layout)
  const int lrow = lane & 15;
  const int b    = blockIdx.x >> 6;     // 64 blocks per batch
  const int m0   = ((blockIdx.x & 63) * 4 + wave) * 16;

  // Q A-fragment (16x32 bf16): lane holds row m0+lrow, k in {0..7,16..23}
  // (half 0) or {8..15,24..31} (half 1) -> two contiguous 16B chunks.
  const __bf16* qrow = qm + ((size_t)b * NN + m0 + lrow) * DD + half * 8;
  const v16bf qa = cat8(*(const v8bf*)qrow, *(const v8bf*)(qrow + 16));

  const __bf16* kbase = km + (size_t)b * NN * DD;
  const v8f zacc = {0.f, 0.f, 0.f, 0.f, 0.f, 0.f, 0.f, 0.f};

  // ---- pass A: row max --------------------------------------------------
  float rmax[8];
#pragma unroll
  for (int r = 0; r < 8; ++r) rmax[r] = -3.0e38f;
  for (int j = 0; j < NN / 16; ++j) {
    const __bf16* krow = kbase + (size_t)(j * 16 + lrow) * DD + half * 8;
    const v16bf kb = cat8(*(const v8bf*)krow, *(const v8bf*)(krow + 16));
    v8f s = __builtin_amdgcn_wmma_f32_16x16x32_bf16(
        false, qa, false, kb, (short)0, zacc, false, false);
#pragma unroll
    for (int r = 0; r < 8; ++r) rmax[r] = fmaxf(rmax[r], s[r]);
  }
#pragma unroll
  for (int r = 0; r < 8; ++r)
#pragma unroll
    for (int off = 1; off < 16; off <<= 1)
      rmax[r] = fmaxf(rmax[r], __shfl_xor(rmax[r], off, 32));

  // ---- pass B: row sum of exp -------------------------------------------
  float rsum[8];
#pragma unroll
  for (int r = 0; r < 8; ++r) rsum[r] = 0.f;
  for (int j = 0; j < NN / 16; ++j) {
    const __bf16* krow = kbase + (size_t)(j * 16 + lrow) * DD + half * 8;
    const v16bf kb = cat8(*(const v8bf*)krow, *(const v8bf*)(krow + 16));
    v8f s = __builtin_amdgcn_wmma_f32_16x16x32_bf16(
        false, qa, false, kb, (short)0, zacc, false, false);
#pragma unroll
    for (int r = 0; r < 8; ++r) rsum[r] += __expf(s[r] - rmax[r]);
  }
  float rinv[8];
#pragma unroll
  for (int r = 0; r < 8; ++r) {
#pragma unroll
    for (int off = 1; off < 16; off <<= 1)
      rsum[r] += __shfl_xor(rsum[r], off, 32);
    rinv[r] = 1.0f / rsum[r];
  }

  // ---- pass C: normalize, write atten, accumulate O = P @ V^T -----------
  v8f o0 = zacc, o1 = zacc;             // d 0..15 and 16..31
  for (int j2 = 0; j2 < NN / 32; ++j2) {
#pragma unroll
    for (int t2 = 0; t2 < 2; ++t2) {
      const int j = j2 * 2 + t2;
      const __bf16* krow = kbase + (size_t)(j * 16 + lrow) * DD + half * 8;
      const v16bf kb = cat8(*(const v8bf*)krow, *(const v8bf*)(krow + 16));
      v8f s = __builtin_amdgcn_wmma_f32_16x16x32_bf16(
          false, qa, false, kb, (short)0, zacc, false, false);
#pragma unroll
      for (int r = 0; r < 8; ++r) {
        const float p = __expf(s[r] - rmax[r]) * rinv[r];
        const int m = r + 8 * half;     // D-matrix row within tile
        atten[((size_t)b * NN + m0 + m) * NN + j * 16 + lrow] = p;
        ldsP[wave][m][t2 * 16 + lrow] = p;   // re-layout via LDS
      }
    }
    // Read P back in A-matrix layout (row lrow, key chunks by half), ->bf16.
    const v4f p0 = *(const v4f*)&ldsP[wave][lrow][half * 8];
    const v4f p1 = *(const v4f*)&ldsP[wave][lrow][half * 8 + 4];
    const v4f p2 = *(const v4f*)&ldsP[wave][lrow][16 + half * 8];
    const v4f p3 = *(const v4f*)&ldsP[wave][lrow][16 + half * 8 + 4];
    v16bf pa;
#pragma unroll
    for (int i = 0; i < 4; ++i) {
      pa[i]      = (__bf16)p0[i];
      pa[4 + i]  = (__bf16)p1[i];
      pa[8 + i]  = (__bf16)p2[i];
      pa[12 + i] = (__bf16)p3[i];
    }
    // V B-fragments: column = d index, K = 32 keys of this chunk.
    const __bf16* v0row = vtm + ((size_t)b * DD + lrow) * NN + j2 * 32 + half * 8;
    const v16bf vb0 = cat8(*(const v8bf*)v0row, *(const v8bf*)(v0row + 16));
    const __bf16* v1row = vtm + ((size_t)b * DD + 16 + lrow) * NN + j2 * 32 + half * 8;
    const v16bf vb1 = cat8(*(const v8bf*)v1row, *(const v8bf*)(v1row + 16));
    o0 = __builtin_amdgcn_wmma_f32_16x16x32_bf16(
        false, pa, false, vb0, (short)0, o0, false, false);
    o1 = __builtin_amdgcn_wmma_f32_16x16x32_bf16(
        false, pa, false, vb1, (short)0, o1, false, false);
  }

  // Store O (fp32 [B,N,32]) for the output projection.
#pragma unroll
  for (int r = 0; r < 8; ++r) {
    const int m = m0 + r + 8 * half;
    obuf[((size_t)b * NN + m) * DD + lrow]      = o0[r];
    obuf[((size_t)b * NN + m) * DD + 16 + lrow] = o1[r];
  }
}

// -------------------------------------------------------------------------
// Kernel 3: out = gamma*(Wo @ O^T + bo) + x   (coalesced over m)
// -------------------------------------------------------------------------
__global__ __launch_bounds__(256) void out_proj(
    const float* __restrict__ obuf, const float* __restrict__ Wo,
    const float* __restrict__ bo, const float* __restrict__ gamma,
    const float* __restrict__ x, float* __restrict__ out)
{
  const int b  = blockIdx.x >> 6;       // 64 blocks per batch
  const int m0 = (blockIdx.x & 63) * 64;
  const int t  = threadIdx.x;
  const int mc = t & 63;
  const int cg = t >> 6;                // 0..3 -> c range cg*64 .. +63
  const int m  = m0 + mc;
  const float g = gamma[0];

  float orow[DD];
  const float* osrc = obuf + ((size_t)b * NN + m) * DD;
#pragma unroll
  for (int d = 0; d < DD; ++d) orow[d] = osrc[d];

  for (int ci = 0; ci < 64; ++ci) {
    const int c = cg * 64 + ci;
    const float* w = Wo + c * DD;
    float acc = 0.f;
#pragma unroll
    for (int d = 0; d < DD; ++d) acc = fmaf(w[d], orow[d], acc);
    const size_t idx = ((size_t)b * CC + c) * NN + m;
    out[idx] = (acc + bo[c]) * g + x[idx];
  }
}

// -------------------------------------------------------------------------
extern "C" void kernel_launch(void* const* d_in, const int* in_sizes, int n_in,
                              void* d_out, int out_size, void* d_ws, size_t ws_size,
                              hipStream_t stream)
{
  const float* x     = (const float*)d_in[0];
  const float* Wq    = (const float*)d_in[1];
  const float* bq    = (const float*)d_in[2];
  const float* Wk    = (const float*)d_in[3];
  const float* bk    = (const float*)d_in[4];
  const float* Wv    = (const float*)d_in[5];
  const float* bv    = (const float*)d_in[6];
  const float* Wo    = (const float*)d_in[7];
  const float* bo    = (const float*)d_in[8];
  const float* gamma = (const float*)d_in[9];

  float* out   = (float*)d_out;
  float* atten = out + (size_t)BB * CC * NN;    // outputs concatenated flat

  char* ws = (char*)d_ws;
  __bf16* qb  = (__bf16*)ws;  ws += (size_t)BB * NN * DD * 2;
  __bf16* kb  = (__bf16*)ws;  ws += (size_t)BB * NN * DD * 2;
  __bf16* vtb = (__bf16*)ws;  ws += (size_t)BB * DD * NN * 2;
  float*  obuf = (float*)ws;                    // BB*NN*DD fp32

  proj_qkv<<<BB * (NN / 32), 256, 0, stream>>>(x, Wq, bq, Wk, bk, Wv, bv,
                                               qb, kb, vtb);
  attn<<<BB * (NN / 64), 128, 0, stream>>>(qb, kb, vtb, atten, obuf);
  out_proj<<<BB * (NN / 64), 256, 0, stream>>>(obuf, Wo, bo, gamma, x, out);
}
